// fusion_23063974379901
// MI455X (gfx1250) — matrix operand order repaired and checked by reading, not compiled, not run on hardware
//
#include <hip/hip_runtime.h>
#include <hip/hip_bf16.h>

// Problem constants from the reference module.
#define N_PTS   1000000
#define W_COLS  70400
#define N_TILES (N_PTS / 16)     // 62500 tiles of 16 points
#define ROWSTR  17               // LDS row stride (floats); 17 keeps B-gather bank-conflict free
#define ACTROWS 64               // padded K range reachable by any B-gather (2 chunks x 32)
#define FILLV   (-9999999.0f)

typedef __attribute__((ext_vector_type(16))) _Float16 v16h;
typedef __attribute__((ext_vector_type(8)))  float    v8f;

static __device__ __forceinline__ v8f wmma_f16(v16h a, v16h b, v8f c) {
    // v_wmma_f32_16x16x32_f16: D = A(16x32 f16) * B(32x16 f16) + C(16x16 f32)
    return __builtin_amdgcn_wmma_f32_16x16x32_f16(false, a, false, b, (short)0, c,
                                                  false, false);
}

// Single-instruction ReLU: v_med3_f32(x, 0, +inf) == clamp-below-at-0.
static __device__ __forceinline__ float relu1(float v) {
    return __builtin_amdgcn_fmed3f(v, 0.0f, __builtin_inff());
}

// Build one 16x32 f16 A-tile of weight matrix Wg[out_ch x in_ch], rows mo..mo+15,
// K-columns ko..ko+31.  Bias is folded in as virtual column K == in_ch (the matching
// activation row is held at 1.0 in the LDS slab).  Padded rows/cols are zero, which
// makes every padded output channel an exact 0 through ReLU -- that is what lets the
// steady-state B-gather run with no predication at all.
// ISA 7.12.2 16-bit A layout: row M = lane%16; element e holds
//   K = ko + e + 8*(lane/16) + (e>=8 ? 8 : 0).
static __device__ __forceinline__
v16h build_A(const float* __restrict__ Wg, const float* __restrict__ Bg,
             int out_ch, int in_ch, int mo, int ko, int lane) {
    int m = mo + (lane & 15);
    int h = lane >> 4;
    v16h a = {};
    if (m < out_ch) {
#pragma unroll
        for (int e = 0; e < 16; ++e) {
            int K = ko + e + 8 * h + ((e >= 8) ? 8 : 0);
            float w = 0.0f;
            if (K < in_ch)       w = Wg[m * in_ch + K];
            else if (K == in_ch) w = Bg[m];
            a[e] = (_Float16)w;
        }
    }
    return a;
}

// Build a 32x16 f16 B-tile from the per-wave LDS activation slab act[row*ROWSTR + col].
// B layout (matches A): col N = lane%16; element e holds
//   K = ko + e + 8*(lane/16) + (e>=8 ? 8 : 0).
// UNCONDITIONAL: the slab invariant guarantees rows [0,64) always hold the correct
// padded value (zeros / folded-bias 1.0), so this is 16 plain ds_load_b32
// (pairable into ds_load_2addr_b32) + 8 v_cvt_pk_f16_f32.
static __device__ __forceinline__
v16h build_B(const float* __restrict__ act, int ko, int lane) {
    int n = lane & 15;
    int h = lane >> 4;
    v16h b = {};
#pragma unroll
    for (int e = 0; e < 16; ++e) {
        int K = ko + e + 8 * h + ((e >= 8) ? 8 : 0);
        b[e] = (_Float16)act[K * ROWSTR + n];
    }
    return b;
}

// Layer-1 B operand straight from global: x is (4, N) row-major; only K<4 is real,
// K==4 is the folded-bias 1.0 row.  All of that lives in the lane/16==0 half.
static __device__ __forceinline__
v16h build_B1(const float* __restrict__ x, int base, int lane) {
    int n = lane & 15;
    int h = lane >> 4;
    v16h b = {};
    if (h == 0) {
#pragma unroll
        for (int c = 0; c < 4; ++c)
            b[c] = (_Float16)x[c * N_PTS + base + n];
        b[4] = (_Float16)1.0f;
    }
    return b;
}

// Store a 16x16 f32 D tile (C/D layout: M = r + 8*(lane/16), N = lane%16) into the
// per-wave activation slab at row offset mo, with optional ReLU.
static __device__ __forceinline__
void store_act(float* __restrict__ act, v8f c, int mo, int lane, bool relu) {
    int n = lane & 15;
    int h = lane >> 4;
#pragma unroll
    for (int r = 0; r < 8; ++r) {
        float v = c[r];
        if (relu) v = relu1(v);
        act[(mo + r + 8 * h) * ROWSTR + n] = v;
    }
}

// Refresh the folded-bias activation row (row = in_ch) with 1.0 after the layer
// whose padded output zeroed it.  One 16-lane predicated store.
static __device__ __forceinline__
void set_bias_row(float* __restrict__ act, int row, int lane) {
    if (lane < 16) act[row * ROWSTR + lane] = 1.0f;
}

// Sign-aware atomic max on float: total IEEE order via int-max for >=0, uint-min for <0.
static __device__ __forceinline__ void atomicMaxF(float* addr, float v) {
    if (v >= 0.0f) atomicMax((int*)addr,          __float_as_int(v));
    else           atomicMin((unsigned int*)addr, __float_as_uint(v));
}

__global__ void init_out_kernel(float* __restrict__ out, int n) {
    int i = blockIdx.x * blockDim.x + threadIdx.x;
    if (i < n) out[i] = (i == n - 1) ? 1.0f : FILLV;  // last slot = flag (scatter branch)
}

__global__ void __launch_bounds__(256)
mlp_scatter_max_kernel(const float* __restrict__ x,
                       const int*   __restrict__ tidx,   // (N,2) [row, col]
                       const float* __restrict__ w1, const float* __restrict__ b1,
                       const float* __restrict__ w2, const float* __restrict__ b2,
                       const float* __restrict__ w3, const float* __restrict__ b3,
                       const float* __restrict__ w4, const float* __restrict__ b4,
                       float* __restrict__ out) {
    __shared__ float act_all[8][ACTROWS * ROWSTR];   // per-wave 64x17 f32 activation slab

    const int lane = threadIdx.x & 31;
    const int wave = threadIdx.x >> 5;
    float* act = act_all[wave];

    // Slab invariant setup: rows 48..63 are never written by any layer store, but the
    // second K-chunk gathers (K = 40 + e + 8h for e>=8) read them -> zero them once.
    // (Rows 0..47 are fully rewritten by L1/L2 stores each tile before being read.)
    {
        int i = lane;                                   // 16 rows x 16 cols = 256 entries
        #pragma unroll
        for (int r = 0; r < 8; ++r, i += 32)
            act[(48 + (i >> 4)) * ROWSTR + (i & 15)] = 0.0f;
    }
    // Per-wave slab, in-order LDS within a wave: no barrier needed.

    // ---- Pre-build all weight A-tiles into VGPRs (13 x v16h), bias folded in. ----
    const v16h a1_0 = build_A(w1, b1, 18,  4,  0,  0, lane);
    const v16h a1_1 = build_A(w1, b1, 18,  4, 16,  0, lane);
    const v16h a2_0 = build_A(w2, b2, 36, 18,  0,  0, lane);
    const v16h a2_1 = build_A(w2, b2, 36, 18, 16,  0, lane);
    const v16h a2_2 = build_A(w2, b2, 36, 18, 32,  0, lane);
    const v16h a3_00 = build_A(w3, b3, 36, 36,  0,  0, lane);
    const v16h a3_01 = build_A(w3, b3, 36, 36,  0, 32, lane);
    const v16h a3_10 = build_A(w3, b3, 36, 36, 16,  0, lane);
    const v16h a3_11 = build_A(w3, b3, 36, 36, 16, 32, lane);
    const v16h a3_20 = build_A(w3, b3, 36, 36, 32,  0, lane);
    const v16h a3_21 = build_A(w3, b3, 36, 36, 32, 32, lane);
    const v16h a4_0  = build_A(w4, b4,  1, 36,  0,  0, lane);
    const v16h a4_1  = build_A(w4, b4,  1, 36,  0, 32, lane);

    const int waveGlobal = blockIdx.x * 8 + wave;
    const int nWaves     = gridDim.x * 8;

    for (int t = waveGlobal; t < N_TILES; t += nWaves) {
        const int base = t * 16;
        v8f c0, c1, c2, zero = {};

        // Hoist the independent scatter-index load; it overlaps the whole MLP and in
        // particular fills L4's WMMA accumulation hazard window.
        int col = -1;
        if ((lane >> 4) == 0)
            col = tidx[2 * (base + lane) + 1];          // tensor_index[p][1]

        // ---- Layer 1: 4 -> 18 (ReLU). B straight from global; writes rows 0..31.
        //      Issue both WMMAs first, then consume (hides WMMA->VALU hazard). ----
        {
            v16h b = build_B1(x, base, lane);
            c0 = wmma_f16(a1_0, b, zero);
            c1 = wmma_f16(a1_1, b, zero);
            store_act(act, c0,  0, lane, true);
            store_act(act, c1, 16, lane, true);
            set_bias_row(act, 18, lane);             // folded-bias row for layer 2
        }
        // ---- Layer 2: 18 -> 36 (ReLU). One K=32 chunk; writes rows 0..47. ----
        {
            v16h b = build_B(act, 0, lane);
            c0 = wmma_f16(a2_0, b, zero);
            c1 = wmma_f16(a2_1, b, zero);
            c2 = wmma_f16(a2_2, b, zero);
            store_act(act, c0,  0, lane, true);
            store_act(act, c1, 16, lane, true);
            store_act(act, c2, 32, lane, true);
            set_bias_row(act, 36, lane);             // folded-bias row for layer 3
        }
        // ---- Layer 3: 36 -> 36 (ReLU). Two K=32 chunks; interleave the three
        //      accumulation pairs so each D->C RAW has >=2 WMMAs in between. ----
        {
            v16h bk0 = build_B(act,  0, lane);
            v16h bk1 = build_B(act, 32, lane);
            c0 = wmma_f16(a3_00, bk0, zero);
            c1 = wmma_f16(a3_10, bk0, zero);
            c2 = wmma_f16(a3_20, bk0, zero);
            c0 = wmma_f16(a3_01, bk1, c0);
            c1 = wmma_f16(a3_11, bk1, c1);
            c2 = wmma_f16(a3_21, bk1, c2);
            store_act(act, c0,  0, lane, true);
            store_act(act, c1, 16, lane, true);
            store_act(act, c2, 32, lane, true);
            set_bias_row(act, 36, lane);             // folded-bias row for layer 4
        }
        // ---- Layer 4: 36 -> 1 (no ReLU). Row M=0 lives in lanes 0..15, c0[0]. ----
        {
            v16h bk0 = build_B(act,  0, lane);
            v16h bk1 = build_B(act, 32, lane);
            c0 = wmma_f16(a4_0, bk0, zero);
            c0 = wmma_f16(a4_1, bk1, c0);
        }

        // ---- Fused scatter + 200-row max: atomic-max into the column bucket. ----
        if (col >= 0 && col < W_COLS)
            atomicMaxF(&out[col], c0[0]);
    }
}

extern "C" void kernel_launch(void* const* d_in, const int* in_sizes, int n_in,
                              void* d_out, int out_size, void* d_ws, size_t ws_size,
                              hipStream_t stream) {
    (void)in_sizes; (void)n_in; (void)d_ws; (void)ws_size;
    const float* x  = (const float*)d_in[0];
    const int*   ti = (const int*)  d_in[1];
    const float* w1 = (const float*)d_in[2]; const float* b1 = (const float*)d_in[3];
    const float* w2 = (const float*)d_in[4]; const float* b2 = (const float*)d_in[5];
    const float* w3 = (const float*)d_in[6]; const float* b3 = (const float*)d_in[7];
    const float* w4 = (const float*)d_in[8]; const float* b4 = (const float*)d_in[9];
    float* out = (float*)d_out;

    // out[0..W-1] = FILL (column max identity), out[last] = flag 1.0
    init_out_kernel<<<(out_size + 255) / 256, 256, 0, stream>>>(out, out_size);

    // 1024 blocks x 8 wave32 = 8192 waves grid-striding over 62500 16-point tiles.
    mlp_scatter_max_kernel<<<1024, 256, 0, stream>>>(
        x, ti, w1, b1, w2, b2, w3, b3, w4, b4, out);
}
